// NetVLAD_83348135346727
// MI455X (gfx1250) — matrix operand order repaired and checked by reading, not compiled, not run on hardware
//
#include <hip/hip_runtime.h>
#include <stdint.h>

#define NN 64
#define CC 512
#define MM 1600
#define KK 64
#define EPSF 1e-12f

typedef __attribute__((ext_vector_type(16))) __bf16 v16bf;
typedef __attribute__((ext_vector_type(8)))  float  v8f;

union V16U { uint4 q[2]; v16bf v; };            // 32B: two 16B vector loads
union V16S { unsigned int u[8]; v16bf v; };     // 32B: gathered packed pairs

__device__ __forceinline__ unsigned short f2bf(float f) {
    union { float f; unsigned int u; } x; x.f = f;
    unsigned int r = x.u + 0x7FFFu + ((x.u >> 16) & 1u);   // round-nearest-even
    return (unsigned short)(r >> 16);
}
__device__ __forceinline__ float bf2f(unsigned short s) {
    union { unsigned int u; float f; } x; x.u = ((unsigned int)s) << 16;
    return x.f;
}

// ---------- K0: conv_w f32 -> bf16 ----------
__global__ void __launch_bounds__(256)
k0_convert_w(const float* __restrict__ w, unsigned short* __restrict__ wbf) {
    int i = blockIdx.x * 256 + threadIdx.x;
    if (i < KK * CC) wbf[i] = f2bf(w[i]);
}

// ---------- K1: per-pixel L2 normalize over C, emit bf16 xn[N][C][M] ----------
__global__ void __launch_bounds__(256)
k1_normalize(const float* __restrict__ x, unsigned short* __restrict__ xn) {
    int p = blockIdx.x * 256 + threadIdx.x;        // p in [0, N*M), exact grid
    int n = p / MM, m = p - n * MM;
    size_t base = (size_t)n * CC * MM + m;
    float s = 0.f;
    for (int c = 0; c < CC; ++c) { float v = x[base + (size_t)c * MM]; s += v * v; }
    float inv = 1.f / fmaxf(sqrtf(s), EPSF);
    for (int c = 0; c < CC; ++c) {                 // second pass is L2-resident
        float v = x[base + (size_t)c * MM];
        xn[base + (size_t)c * MM] = f2bf(v * inv);
    }
}

// ---------- K2: logits via WMMA + softmax over K ----------
// Block = 4 waves x 16 pixels; wave computes D(k,m) tiles for all 64 k.
__global__ void __launch_bounds__(128)
k2_logits_softmax(const unsigned short* __restrict__ xn,
                  const unsigned short* __restrict__ wbf,
                  const float* __restrict__ bias,
                  unsigned short* __restrict__ a) {
    __shared__ float sL[4][16][65];                // [wave][pixel][k], padded
    int tid = threadIdx.x;
    int wv = tid >> 5, lane = tid & 31;
    int l16 = lane & 15, hi = lane >> 4;
    int n  = blockIdx.x / 25;
    int m0 = (blockIdx.x % 25) * 64 + wv * 16;

    const unsigned short* xbase = xn + (size_t)n * CC * MM;
    v8f zero = {0.f,0.f,0.f,0.f,0.f,0.f,0.f,0.f};
    v8f acc[4] = {zero, zero, zero, zero};

    for (int c0 = 0; c0 < CC; c0 += 32) {
        // B tile (K=c rows, N=m cols): lane = column m0+l16; half h takes c 8h..8h+7 and 16+8h..16+8h+7
        V16S bb;
        const unsigned short* xp = xbase + m0 + l16;
        int cb = c0 + hi * 8;
        #pragma unroll
        for (int v = 0; v < 8; ++v) {
            unsigned int lo = xp[(size_t)(cb + 2 * v + (v & 4) * 2) * MM];
            unsigned int hh = xp[(size_t)(cb + 2 * v + (v & 4) * 2 + 1) * MM];
            bb.u[v] = lo | (hh << 16);
        }
        #pragma unroll
        for (int kt = 0; kt < 4; ++kt) {
            // A tile (M=k rows, K=c): lane = row; per-lane two 16B contiguous chunks
            const unsigned short* wr = wbf + (size_t)(kt * 16 + l16) * CC + c0 + hi * 8;
            V16U aa;
            aa.q[0] = *(const uint4*)(wr);
            aa.q[1] = *(const uint4*)(wr + 16);
            acc[kt] = __builtin_amdgcn_wmma_f32_16x16x32_bf16(
                false, aa.v, false, bb.v, (short)0, acc[kt], false, false);
        }
    }
    // D layout: col m = l16, row k = kt*16 + r + hi*8  -> dump to LDS
    #pragma unroll
    for (int kt = 0; kt < 4; ++kt)
        #pragma unroll
        for (int r = 0; r < 8; ++r)
            sL[wv][l16][kt * 16 + r + hi * 8] = acc[kt][r];
    __syncthreads();

    // stable softmax over K=64: lane l16 = pixel, hi selects k-half; combine halves via shfl
    float vreg[32];
    float mx = -3.4e38f;
    #pragma unroll
    for (int j = 0; j < 32; ++j) {
        int k = hi * 32 + j;
        float v = sL[wv][l16][k] + bias[k];
        vreg[j] = v;
        mx = fmaxf(mx, v);
    }
    mx = fmaxf(mx, __shfl_xor(mx, 16, 32));
    float sum = 0.f;
    #pragma unroll
    for (int j = 0; j < 32; ++j) { vreg[j] = expf(vreg[j] - mx); sum += vreg[j]; }
    sum += __shfl_xor(sum, 16, 32);
    float inv = 1.f / sum;
    unsigned short* ap = a + (size_t)n * KK * MM + m0 + l16;
    #pragma unroll
    for (int j = 0; j < 32; ++j) {
        int k = hi * 32 + j;
        ap[(size_t)k * MM] = f2bf(vreg[j] * inv);
    }
}

// ---------- K3b: a_sum[n][k] = sum_m a[n][k][m] (one wave per row) ----------
__global__ void __launch_bounds__(256)
k3b_asum(const unsigned short* __restrict__ a, float* __restrict__ a_sum) {
    int wv = threadIdx.x >> 5, lane = threadIdx.x & 31;
    int row = blockIdx.x * 8 + wv;                 // row < N*K = 4096, exact grid
    const unsigned short* ap = a + (size_t)row * MM;
    float s = 0.f;
    for (int i = 0; i < MM / 32; ++i) s += bf2f(ap[i * 32 + lane]);
    #pragma unroll
    for (int o = 16; o > 0; o >>= 1) s += __shfl_xor(s, o, 32);
    if (lane == 0) a_sum[row] = s;
}

// ---------- K3: aggregation GEMM via WMMA, D(c,k) = sum_m xn[c][m] * a[k][m] ----------
// Both A and B are row-contiguous 16B loads. Store aggT[N][C][K].
__global__ void __launch_bounds__(256)
k3_agg(const unsigned short* __restrict__ xn, const unsigned short* __restrict__ a,
       float* __restrict__ aggT) {
    int tid = threadIdx.x;
    int wv = tid >> 5, lane = tid & 31;
    int l16 = lane & 15, hi = lane >> 4;
    int n  = blockIdx.x >> 2;
    int c0 = (blockIdx.x & 3) * 128 + wv * 16;

    const unsigned short* xrow = xn + ((size_t)n * CC + (c0 + l16)) * MM;
    v8f zero = {0.f,0.f,0.f,0.f,0.f,0.f,0.f,0.f};
    v8f acc[4] = {zero, zero, zero, zero};

    for (int m0 = 0; m0 < MM; m0 += 32) {
        // A tile (M=c rows, K=m): lane = row c0+l16
        const unsigned short* xp = xrow + m0 + hi * 8;
        __builtin_prefetch(xp + 32, 0, 1);         // global_prefetch_b8 next chunk
        V16U aa;
        aa.q[0] = *(const uint4*)(xp);
        aa.q[1] = *(const uint4*)(xp + 16);
        #pragma unroll
        for (int kt = 0; kt < 4; ++kt) {
            // B tile (K=m rows, N=k cols): lane = col kt*16+l16, m split by half-wave
            const unsigned short* ar = a + ((size_t)n * KK + (kt * 16 + l16)) * MM
                                         + m0 + hi * 8;
            V16U bb;
            bb.q[0] = *(const uint4*)(ar);
            bb.q[1] = *(const uint4*)(ar + 16);
            acc[kt] = __builtin_amdgcn_wmma_f32_16x16x32_bf16(
                false, aa.v, false, bb.v, (short)0, acc[kt], false, false);
        }
    }
    // D: col k = kt*16+l16, row c = c0 + r + hi*8 ; each half-wave gives 16 contiguous k stores
    #pragma unroll
    for (int kt = 0; kt < 4; ++kt)
        #pragma unroll
        for (int r = 0; r < 8; ++r) {
            int c = c0 + r + hi * 8;
            int k = kt * 16 + l16;
            aggT[((size_t)n * CC + c) * KK + k] = acc[kt][r];
        }
}

// ---------- K4: vlad = aggT^T - a_sum*cent, intra-norm, global norm ----------
__global__ void __launch_bounds__(256)
k4_finalize(const float* __restrict__ aggT, const float* __restrict__ a_sum,
            const float* __restrict__ cent, float* __restrict__ out) {
    __shared__ float ssum[64];
    __shared__ float sinv[64];
    __shared__ float wsum[8];
    __shared__ float ginvs;
    int n = blockIdx.x;
    int tid = threadIdx.x, wv = tid >> 5, lane = tid & 31;
    int c1 = tid, c2 = tid + 256;
    const float* ag = aggT + (size_t)n * CC * KK;
    const float* as = a_sum + n * KK;
    float* op = out + (size_t)n * KK * CC;

    for (int k = 0; k < KK; ++k) {
        float sk = as[k];
        float v1 = ag[(size_t)c1 * KK + k] - sk * cent[k * CC + c1];
        float v2 = ag[(size_t)c2 * KK + k] - sk * cent[k * CC + c2];
        float ps = v1 * v1 + v2 * v2;
        #pragma unroll
        for (int o = 16; o > 0; o >>= 1) ps += __shfl_xor(ps, o, 32);
        if (lane == 0) wsum[wv] = ps;
        __syncthreads();
        if (tid == 0) {
            float s = 0.f;
            for (int i = 0; i < 8; ++i) s += wsum[i];
            ssum[k] = s;
            sinv[k] = 1.f / fmaxf(sqrtf(s), EPSF);
        }
        __syncthreads();
        float iv = sinv[k];
        op[(size_t)k * CC + c1] = v1 * iv;
        op[(size_t)k * CC + c2] = v2 * iv;
    }
    __syncthreads();
    if (tid == 0) {
        float g = 0.f;
        for (int k = 0; k < KK; ++k) g += ssum[k] * sinv[k] * sinv[k];
        ginvs = 1.f / fmaxf(sqrtf(g), EPSF);
    }
    __syncthreads();
    float gi = ginvs;
    for (int k = 0; k < KK; ++k) {      // rescale own elements (thread-private RMW)
        op[(size_t)k * CC + c1] *= gi;
        op[(size_t)k * CC + c2] *= gi;
    }
}

extern "C" void kernel_launch(void* const* d_in, const int* in_sizes, int n_in,
                              void* d_out, int out_size, void* d_ws, size_t ws_size,
                              hipStream_t stream) {
    (void)in_sizes; (void)n_in; (void)out_size; (void)ws_size;
    const float* x      = (const float*)d_in[0];   // [N,C,H,W]
    const float* conv_w = (const float*)d_in[1];   // [K,C]
    const float* conv_b = (const float*)d_in[2];   // [K]
    const float* cent   = (const float*)d_in[3];   // [K,C]
    float* out = (float*)d_out;                    // [N, K*C]

    char* ws = (char*)d_ws;
    // workspace layout (all 16B aligned), total ~126.4 MB
    unsigned short* xn   = (unsigned short*)(ws);                // 104,857,600 B
    unsigned short* wbf  = (unsigned short*)(ws + 104857600);    //      65,536 B
    unsigned short* asgn = (unsigned short*)(ws + 104923136);    //  13,107,200 B
    float*          aggT = (float*)         (ws + 118030336);    //   8,388,608 B
    float*          asum = (float*)         (ws + 126418944);    //      16,384 B

    k0_convert_w     <<<128,      256, 0, stream>>>(conv_w, wbf);
    k1_normalize     <<<400,      256, 0, stream>>>(x, xn);
    k2_logits_softmax<<<NN * 25,  128, 0, stream>>>(xn, wbf, conv_b, asgn);
    k3b_asum         <<<512,      256, 0, stream>>>(asgn, asum);
    k3_agg           <<<NN * 4,   256, 0, stream>>>(xn, asgn, aggT);
    k4_finalize      <<<NN,       256, 0, stream>>>(aggT, asum, cent, out);
}
